// Sea_Attention_14723147890796
// MI455X (gfx1250) — matrix operand order repaired and checked by reading, not compiled, not run on hardware
//
#include <hip/hip_runtime.h>
#include <cstdint>
#include <cstddef>

// ---------------------------------------------------------------------------
// Sea_Attention forward for MI455X (gfx1250, wave32, WMMA).
// Every conv(+BN) is an f16 WMMA GEMM (v_wmma_f32_16x16x32_f16, f32 accum),
// BN folded into weights. Dense 3x3 convs = implicit GEMM (K = Cin*9).
// A tiles staged via GLOBAL_LOAD_ASYNC_TO_LDS_B128 (ASYNCcnt) when available.
// B tiles staged transposed so fragments are 2x ds_load_b128 per tile.
// ---------------------------------------------------------------------------

typedef _Float16 h16;
typedef __attribute__((ext_vector_type(16))) _Float16 v16h;
typedef __attribute__((ext_vector_type(8)))  _Float16 v8h;
typedef __attribute__((ext_vector_type(8)))  float    v8f;
typedef int v4i __attribute__((vector_size(16)));

#define BB   8
#define HH   64
#define WW   64
#define HWP  4096
#define SCALE_ATTN 0.17677669529663687f   // 32^-0.5

#define LDA 40   // padded LDS row stride (halves) for A tile (128 x 32)
#define LDT 40   // padded LDS row stride (halves) for B^T tile (64 cols x 32 K)

#ifndef __has_builtin
#define __has_builtin(x) 0
#endif
#if __has_builtin(__builtin_amdgcn_global_load_async_to_lds_b128) && \
    __has_builtin(__builtin_amdgcn_s_wait_asynccnt)
#define USE_ASYNC_LDS 1
#else
#define USE_ASYNC_LDS 0
#endif

#if USE_ASYNC_LDS
typedef __attribute__((address_space(1))) v4i* gas_b128;
typedef __attribute__((address_space(3))) v4i* las_b128;
__device__ __forceinline__ void async_g2l_b128(const h16* g, h16* l)
{
    __builtin_amdgcn_global_load_async_to_lds_b128(
        (gas_b128)(void*)g, (las_b128)(void*)l, 0, 0);
}
__device__ __forceinline__ void async_wait0() { __builtin_amdgcn_s_wait_asynccnt(0); }
#else
__device__ __forceinline__ void async_wait0() {}
#endif

// ---------------------------------------------------------------------------
// Fold BN into conv weights.
// ---------------------------------------------------------------------------
__global__ void prep_weights(const float* __restrict__ w,
                             const float* __restrict__ g,
                             const float* __restrict__ bb,
                             const float* __restrict__ m,
                             const float* __restrict__ v,
                             float eps, int Co, int Kw,
                             h16* __restrict__ w16,
                             float* __restrict__ bias)
{
    int i = blockIdx.x * blockDim.x + threadIdx.x;
    int total = Co * Kw;
    if (i < total) {
        int o = i / Kw;
        float s = g[o] * rsqrtf(v[o] + eps);
        w16[i] = (h16)(w[i] * s);
    }
    if (i < Co) {
        float s = g[i] * rsqrtf(v[i] + eps);
        bias[i] = bb[i] - m[i] * s;
    }
}

// ---------------------------------------------------------------------------
// Fragment loads: both A row and B^T column are 2 contiguous 16B LDS chunks.
// ---------------------------------------------------------------------------
__device__ __forceinline__ v16h frag_from_2x8(const h16* p0, const h16* p1)
{
    v8h lo = *(const v8h*)p0;
    v8h hi = *(const v8h*)p1;
    return __builtin_shufflevector(lo, hi, 0,1,2,3,4,5,6,7,8,9,10,11,12,13,14,15);
}

template<int ACT, int COMB>
__device__ __forceinline__ void epi(
    float accv, int b, int co, int col, int Co, int P,
    const float* __restrict__ bias, const float* __restrict__ act_a,
    h16* __restrict__ Out16, float* __restrict__ Out32,
    const h16* __restrict__ Add16, const float* __restrict__ Gate,
    int outTotC, int outChOff)
{
    float vv = accv + bias[co];
    if (ACT == 1)      vv = vv > 0.f ? vv : 0.f;
    else if (ACT == 2) vv = vv >= 0.f ? vv : act_a[co] * vv;
    else if (ACT == 3) vv = vv >= 0.f ? vv : act_a[0] * vv;
    const size_t sidx = ((size_t)b * Co + co) * (size_t)P + col;
    const size_t oidx = ((size_t)b * outTotC + outChOff + co) * (size_t)P + col;
    if (COMB == 0)      Out16[oidx] = (h16)vv;
    else if (COMB == 1) Out16[oidx] = (h16)(vv + (float)Add16[sidx]);
    else if (COMB == 2) Out32[sidx] = vv;
    else {
        float g = vv + 3.f;
        g = g < 0.f ? 0.f : (g > 6.f ? 6.f : g);
        Out32[sidx] = (g * (1.f / 6.f)) * Gate[sidx];
    }
}

// ---------------------------------------------------------------------------
// 1x1 conv (+folded BN) GEMM. Block = 256 threads (8 waves, wave32),
// 128x64 output tile, K-steps of 32, waves arranged 4(M) x 2(N), each 2x2 WMMA.
// ---------------------------------------------------------------------------
template<int IN32, int ACT, int COMB>
__global__ __launch_bounds__(256)
void gemm1x1_wmma(const h16* __restrict__ W,
                  const h16* __restrict__ In16,
                  const float* __restrict__ In32,
                  const float* __restrict__ bias,
                  const float* __restrict__ act_a,
                  h16* __restrict__ Out16, float* __restrict__ Out32,
                  const h16* __restrict__ Add16, const float* __restrict__ Gate,
                  int Co, int K, int P, int outTotC, int outChOff)
{
    __shared__ h16 As[128 * LDA];   // A tile 128 x 32, row-major
    __shared__ h16 BsT[64 * LDT];   // B tile transposed: [pixel][K]
    const int b    = blockIdx.z;
    const int m0   = blockIdx.y * 128;
    const int n0   = blockIdx.x * 64;
    const int t    = threadIdx.x;
    const int lane = t & 31;
    const int wave = t >> 5;
    const int wm   = (wave >> 1) * 32;
    const int wn   = (wave & 1) * 32;
    const int ml   = lane & 15;
    const int kh   = lane >> 4;

    v8f acc[2][2] = {};

    const h16*   inb16 = IN32 ? nullptr : In16 + (size_t)b * K * P;
    const float* inb32 = IN32 ? In32 + (size_t)b * K * P : nullptr;

    const int arow = t >> 1;          // A row staged by this thread (0..127)
    const int acb  = (t & 1) * 16;    // A col base
    const int brow = t >> 3;          // B K-row staged (0..31)
    const int bcb  = (t & 7) * 8;     // B pixel base (8 pixels per thread)

    for (int k0 = 0; k0 < K; k0 += 32) {
        { // stage A (BN-folded f16 weights), 2 x 16B per thread
            const h16* src = W + (size_t)(m0 + arow) * K + (k0 + acb);
            h16* dst = &As[arow * LDA + acb];
#if USE_ASYNC_LDS
            async_g2l_b128(src, dst);
            async_g2l_b128(src + 8, dst + 8);
#else
            *(v8h*)dst       = *(const v8h*)src;
            *(v8h*)(dst + 8) = *(const v8h*)(src + 8);
#endif
            if (k0 + 32 < K) __builtin_prefetch(src + 32, 0, 1);
        }
        { // stage B transposed: thread reads 8 pixels of one channel row,
          // scatters into BsT[pixel][K]
            if (IN32) {
                const float* s = inb32 + (size_t)(k0 + brow) * P + (n0 + bcb);
#pragma unroll
                for (int j = 0; j < 8; j++)
                    BsT[(bcb + j) * LDT + brow] = (h16)s[j];
            } else {
                const h16* s = inb16 + (size_t)(k0 + brow) * P + (n0 + bcb);
#pragma unroll
                for (int j = 0; j < 8; j++)
                    BsT[(bcb + j) * LDT + brow] = s[j];
            }
        }
        async_wait0();
        __syncthreads();
        v16h af[2], bf[2];
#pragma unroll
        for (int mi = 0; mi < 2; mi++) {
            const h16* ar = &As[(wm + mi * 16 + ml) * LDA];
            af[mi] = frag_from_2x8(ar + kh * 8, ar + 16 + kh * 8);
        }
#pragma unroll
        for (int ni = 0; ni < 2; ni++) {
            const h16* bc = &BsT[(wn + ni * 16 + ml) * LDT];
            bf[ni] = frag_from_2x8(bc + kh * 8, bc + 16 + kh * 8);
        }
#pragma unroll
        for (int mi = 0; mi < 2; mi++)
#pragma unroll
            for (int ni = 0; ni < 2; ni++)
                acc[mi][ni] = __builtin_amdgcn_wmma_f32_16x16x32_f16(
                    false, af[mi], false, bf[ni], (short)0, acc[mi][ni], false, false);
        __syncthreads();
    }

#pragma unroll
    for (int mi = 0; mi < 2; mi++)
#pragma unroll
        for (int ni = 0; ni < 2; ni++) {
            const int col = n0 + wn + ni * 16 + ml;
#pragma unroll
            for (int r = 0; r < 8; r++) {
                const int co = m0 + wm + mi * 16 + r + 8 * kh;
                epi<ACT, COMB>(acc[mi][ni][r], b, co, col, Co, P,
                               bias, act_a, Out16, Out32, Add16, Gate,
                               outTotC, outChOff);
            }
        }
}

// ---------------------------------------------------------------------------
// 3x3 conv (pad=1) as implicit GEMM: K = Cin*9, im2col gather into BsT.
// ---------------------------------------------------------------------------
template<int ACT, int COMB>
__global__ __launch_bounds__(256)
void gemm3x3_wmma(const h16* __restrict__ W,        // Co x (Cin*9), OIHW
                  const h16* __restrict__ In16,     // (B, Cin, 64, 64)
                  const float* __restrict__ bias,
                  const float* __restrict__ act_a,
                  h16* __restrict__ Out16,
                  const h16* __restrict__ Add16,
                  int Co, int Cin, int outTotC, int outChOff)
{
    __shared__ h16 As[128 * LDA];
    __shared__ h16 BsT[64 * LDT];
    const int K    = Cin * 9;
    const int P    = HWP;
    const int b    = blockIdx.z;
    const int m0   = blockIdx.y * 128;
    const int n0   = blockIdx.x * 64;
    const int t    = threadIdx.x;
    const int lane = t & 31;
    const int wave = t >> 5;
    const int wm   = (wave >> 1) * 32;
    const int wn   = (wave & 1) * 32;
    const int ml   = lane & 15;
    const int kh   = lane >> 4;

    v8f acc[2][2] = {};
    const h16* inb = In16 + (size_t)b * Cin * HWP;

    const int arow = t >> 1;
    const int acb  = (t & 1) * 16;
    const int brow = t >> 3;
    const int bcb  = (t & 7) * 8;

    // incremental kk = k0 + brow -> (ci, rem); +32 per step: ci+=3, rem+=5
    int ci  = brow / 9;
    int rem = brow - ci * 9;

    for (int k0 = 0; k0 < K; k0 += 32) {
        { // A
            const h16* src = W + (size_t)(m0 + arow) * K + (k0 + acb);
            h16* dst = &As[arow * LDA + acb];
#if USE_ASYNC_LDS
            async_g2l_b128(src, dst);
            async_g2l_b128(src + 8, dst + 8);
#else
            *(v8h*)dst       = *(const v8h*)src;
            *(v8h*)(dst + 8) = *(const v8h*)(src + 8);
#endif
            if (k0 + 32 < K) __builtin_prefetch(src + 32, 0, 1);
        }
        { // B: im2col gather with zero pad, transposed into BsT
            const int dy = rem / 3 - 1;
            const int dx = rem - (rem / 3) * 3 - 1;
            const h16* cbase = inb + (size_t)ci * HWP;
#pragma unroll
            for (int j = 0; j < 8; j++) {
                const int p = n0 + bcb + j;
                const int y = (p >> 6) + dy;
                const int x = (p & 63) + dx;
                h16 val = (h16)0.f;
                if (y >= 0 && y < HH && x >= 0 && x < WW)
                    val = cbase[y * WW + x];
                BsT[(bcb + j) * LDT + brow] = val;
            }
            ci += 3; rem += 5;
            if (rem >= 9) { rem -= 9; ci += 1; }
        }
        async_wait0();
        __syncthreads();
        v16h af[2], bf[2];
#pragma unroll
        for (int mi = 0; mi < 2; mi++) {
            const h16* ar = &As[(wm + mi * 16 + ml) * LDA];
            af[mi] = frag_from_2x8(ar + kh * 8, ar + 16 + kh * 8);
        }
#pragma unroll
        for (int ni = 0; ni < 2; ni++) {
            const h16* bc = &BsT[(wn + ni * 16 + ml) * LDT];
            bf[ni] = frag_from_2x8(bc + kh * 8, bc + 16 + kh * 8);
        }
#pragma unroll
        for (int mi = 0; mi < 2; mi++)
#pragma unroll
            for (int ni = 0; ni < 2; ni++)
                acc[mi][ni] = __builtin_amdgcn_wmma_f32_16x16x32_f16(
                    false, af[mi], false, bf[ni], (short)0, acc[mi][ni], false, false);
        __syncthreads();
    }

#pragma unroll
    for (int mi = 0; mi < 2; mi++)
#pragma unroll
        for (int ni = 0; ni < 2; ni++) {
            const int col = n0 + wn + ni * 16 + ml;
#pragma unroll
            for (int r = 0; r < 8; r++) {
                const int co = m0 + wm + mi * 16 + r + 8 * kh;
                epi<ACT, COMB>(acc[mi][ni][r], b, co, col, Co, P,
                               bias, act_a, Out16, nullptr, Add16, nullptr,
                               outTotC, outChOff);
            }
        }
}

// ---------------------------------------------------------------------------
// Fused depthwise branch (dconv + ddconv + bp) -> QKV2[:, 512:, :]
// ---------------------------------------------------------------------------
__global__ void dw_fuse(const h16* __restrict__ BR,
    const float* __restrict__ wd,  const float* __restrict__ gd,
    const float* __restrict__ bd,  const float* __restrict__ md,
    const float* __restrict__ vd,  const float* __restrict__ ad,
    const float* __restrict__ wdd, const float* __restrict__ gdd,
    const float* __restrict__ bdd, const float* __restrict__ mdd,
    const float* __restrict__ vdd, const float* __restrict__ add_,
    const float* __restrict__ gbp, const float* __restrict__ bbp,
    const float* __restrict__ mbp, const float* __restrict__ vbp,
    const float* __restrict__ abp,
    h16* __restrict__ Out, int outTotC, int outChOff)
{
    const size_t i = (size_t)blockIdx.x * blockDim.x + threadIdx.x;
    const size_t total = (size_t)BB * 512 * HWP;
    if (i >= total) return;
    const int p = (int)(i & (HWP - 1));
    const int c = (int)((i >> 12) & 511);
    const int b = (int)(i >> 21);
    const int y = p >> 6, x = p & 63;
    const h16* base = BR + ((size_t)b * 512 + c) * HWP;
    const float br = (float)base[p];
    float s1 = 0.f, s2 = 0.f;
#pragma unroll
    for (int ky = 0; ky < 3; ky++)
#pragma unroll
        for (int kx = 0; kx < 3; kx++) {
            const float w1 = wd[c * 9 + ky * 3 + kx];
            const float w2 = wdd[c * 9 + ky * 3 + kx];
            int y1 = y + ky - 1, x1 = x + kx - 1;
            if (y1 >= 0 && y1 < HH && x1 >= 0 && x1 < WW)
                s1 += w1 * (float)base[y1 * WW + x1];
            int y2 = y + (ky - 1) * 2, x2 = x + (kx - 1) * 2;
            if (y2 >= 0 && y2 < HH && x2 >= 0 && x2 < WW)
                s2 += w2 * (float)base[y2 * WW + x2];
        }
    float sc1 = gd[c] * rsqrtf(vd[c] + 1e-3f);
    float br1 = sc1 * s1 + (bd[c] - md[c] * sc1);
    br1 = br1 >= 0.f ? br1 : ad[c] * br1;
    float sc2 = gdd[c] * rsqrtf(vdd[c] + 1e-3f);
    float br2 = sc2 * s2 + (bdd[c] - mdd[c] * sc2);
    br2 = br2 >= 0.f ? br2 : add_[c] * br2;
    float s = br + br1 + br2;
    float scb = gbp[c] * rsqrtf(vbp[c] + 1e-3f);
    float o = scb * s + (bbp[c] - mbp[c] * scb);
    o = o >= 0.f ? o : abp[c] * o;
    Out[((size_t)b * outTotC + outChOff + c) * HWP + p] = (h16)o;
}

// ---------------------------------------------------------------------------
// Axial means over W (row) and H (col)
// ---------------------------------------------------------------------------
__global__ void pool_means(const h16* __restrict__ src, int C,
                           float* __restrict__ rowOut, float* __restrict__ colOut)
{
    const int i = blockIdx.x * blockDim.x + threadIdx.x;
    const int total = BB * C * 64;
    if (i >= total) return;
    const int n = i & 63;
    const int c = (i >> 6) % C;
    const int b = i / (C * 64);
    const h16* base = src + ((size_t)b * C + c) * HWP;
    float sr = 0.f, sc = 0.f;
#pragma unroll 8
    for (int j = 0; j < 64; j++) {
        sr += (float)base[n * 64 + j];
        sc += (float)base[j * 64 + n];
    }
    rowOut[i] = sr * (1.f / 64.f);
    colOut[i] = sc * (1.f / 64.f);
}

// ---------------------------------------------------------------------------
// Axial attention per (b, head); softmax((q+peq)(k+pek)^T * scale) @ v, relu.
// ---------------------------------------------------------------------------
__global__ __launch_bounds__(64)
void axial_attn(const float* __restrict__ qkP,  // (b,512,64): q ch 0-255, k 256-511
                const float* __restrict__ vP,   // (b,512,64)
                const float* __restrict__ peq,  // (256,16)
                const float* __restrict__ pek,  // (256,16)
                h16* __restrict__ out16)        // (b,512,64)
{
    __shared__ float qs[64][33];
    __shared__ float ks[32][65];
    __shared__ float vs[64][65];
    const int h = blockIdx.x;
    const int b = blockIdx.y;
    const int n = threadIdx.x;

    float pos = (n + 0.5f) * (16.f / 64.f) - 0.5f;
    pos = fminf(fmaxf(pos, 0.f), 15.f);
    const int lo = (int)floorf(pos);
    const int hi = lo + 1 > 15 ? 15 : lo + 1;
    const float tt = pos - (float)lo;

    for (int d = 0; d < 32; d++) {
        const int cq = h * 32 + d;
        const float pq = peq[cq * 16 + lo] * (1.f - tt) + peq[cq * 16 + hi] * tt;
        const float pk = pek[cq * 16 + lo] * (1.f - tt) + pek[cq * 16 + hi] * tt;
        qs[n][d] = qkP[((size_t)b * 512 + cq) * 64 + n] + pq;
        ks[d][n] = qkP[((size_t)b * 512 + 256 + cq) * 64 + n] + pk;
    }
    for (int d = 0; d < 64; d++)
        vs[n][d] = vP[((size_t)b * 512 + h * 64 + d) * 64 + n];
    __syncthreads();

    float logits[64];
    float mx = -1e30f;
    for (int m = 0; m < 64; m++) {
        float s = 0.f;
        for (int d = 0; d < 32; d++) s += qs[n][d] * ks[d][m];
        s *= SCALE_ATTN;
        logits[m] = s;
        mx = fmaxf(mx, s);
    }
    float den = 0.f;
    for (int m = 0; m < 64; m++) { logits[m] = __expf(logits[m] - mx); den += logits[m]; }
    const float inv = 1.f / den;
    for (int d = 0; d < 64; d++) {
        float o = 0.f;
        for (int m = 0; m < 64; m++) o += logits[m] * vs[m][d];
        o *= inv;
        o = o > 0.f ? o : 0.f;
        out16[((size_t)b * 512 + h * 64 + d) * 64 + n] = (h16)o;
    }
}

// ---------------------------------------------------------------------------
// xx = relu(v + xx_row[y] + xx_col[x]) -> f16 for final proj GEMM
// ---------------------------------------------------------------------------
__global__ void xx_combine(const h16* __restrict__ V,
                           const float* __restrict__ PR,
                           const float* __restrict__ PC,
                           h16* __restrict__ XX)
{
    const size_t i = (size_t)blockIdx.x * blockDim.x + threadIdx.x;
    const size_t total = (size_t)BB * 512 * HWP;
    if (i >= total) return;
    const int p = (int)(i & (HWP - 1));
    const int c = (int)((i >> 12) & 511);
    const int b = (int)(i >> 21);
    const int y = p >> 6, x = p & 63;
    float vv = (float)V[i] + PR[((size_t)b * 512 + c) * 64 + y]
                           + PC[((size_t)b * 512 + c) * 64 + x];
    XX[i] = (h16)(vv > 0.f ? vv : 0.f);
}

// ---------------------------------------------------------------------------
// Host launcher
// ---------------------------------------------------------------------------
static inline void* wsp(void* ws, size_t& off, size_t bytes)
{
    void* p = (char*)ws + off;
    off += (bytes + 255) & ~(size_t)255;
    return p;
}

extern "C" void kernel_launch(void* const* d_in, const int* in_sizes, int n_in,
                              void* d_out, int out_size, void* d_ws, size_t ws_size,
                              hipStream_t stream)
{
    (void)in_sizes; (void)n_in; (void)out_size; (void)ws_size;
#define FIN(i) ((const float*)d_in[i])
    // Input order: x first, then params flattened as a jax pytree
    // (dicts in sorted-key order; bn dicts: b,g,m,v).
    const float* x = FIN(0);
    const float *l1_a = FIN(1), *l1_b = FIN(2), *l1_g = FIN(3), *l1_m = FIN(4), *l1_v = FIN(5), *l1_w = FIN(6);
    const float *l3_a = FIN(7), *l3_b = FIN(8), *l3_g = FIN(9), *l3_m = FIN(10), *l3_v = FIN(11), *l3_w = FIN(12);
    const float *bp_a = FIN(13), *bp_b = FIN(14), *bp_g = FIN(15), *bp_m = FIN(16), *bp_v = FIN(17);
    const float *c3_a = FIN(18), *c3_b = FIN(19), *c3_g = FIN(20), *c3_m = FIN(21), *c3_v = FIN(22), *c3_w = FIN(23);
    const float *dc_a = FIN(24), *dc_b = FIN(25), *dc_g = FIN(26), *dc_m = FIN(27), *dc_v = FIN(28), *dc_w = FIN(29);
    const float *dd_a = FIN(30), *dd_b = FIN(31), *dd_g = FIN(32), *dd_m = FIN(33), *dd_v = FIN(34), *dd_w = FIN(35);
    const float *pe_colk = FIN(36), *pe_colq = FIN(37), *pe_rowk = FIN(38), *pe_rowq = FIN(39);
    const float *pj_b = FIN(40), *pj_g = FIN(41), *pj_m = FIN(42), *pj_v = FIN(43), *pj_w = FIN(44);
    const float *pc_b = FIN(45), *pc_g = FIN(46), *pc_m = FIN(47), *pc_v = FIN(48), *pc_w = FIN(49);
    const float *pr_b = FIN(50), *pr_g = FIN(51), *pr_m = FIN(52), *pr_v = FIN(53), *pr_w = FIN(54);
    const float *pw_b = FIN(55), *pw_g = FIN(56), *pw_m = FIN(57), *pw_v = FIN(58), *pw_w = FIN(59);
    const float *tk_b = FIN(60), *tk_g = FIN(61), *tk_m = FIN(62), *tk_v = FIN(63), *tk_w = FIN(64);
    const float *tq_b = FIN(65), *tq_g = FIN(66), *tq_m = FIN(67), *tq_v = FIN(68), *tq_w = FIN(69);
    const float *tv_b = FIN(70), *tv_g = FIN(71), *tv_m = FIN(72), *tv_v = FIN(73), *tv_w = FIN(74);

    size_t off = 0;
    h16*   QK16  = (h16*)  wsp(d_ws, off, (size_t)BB*512*HWP*2);   // q ch 0-255, k ch 256-511
    h16*   V16   = (h16*)  wsp(d_ws, off, (size_t)BB*512*HWP*2);
    h16*   BR16  = (h16*)  wsp(d_ws, off, (size_t)BB*512*HWP*2);   // conv3x3 out; reused as XX
    h16*   QKV2  = (h16*)  wsp(d_ws, off, (size_t)BB*1024*HWP*2);  // [x1 ; x2]
    h16*   T16   = (h16*)  wsp(d_ws, off, (size_t)BB*512*HWP*2);   // left1 branch
    float* PW    = (float*)wsp(d_ws, off, (size_t)BB*256*HWP*4);   // pwconv out (gated qkv)
    float* QKROW = (float*)wsp(d_ws, off, (size_t)BB*512*64*4);
    float* QKCOL = (float*)wsp(d_ws, off, (size_t)BB*512*64*4);
    float* VROW  = (float*)wsp(d_ws, off, (size_t)BB*512*64*4);
    float* VCOL  = (float*)wsp(d_ws, off, (size_t)BB*512*64*4);
    h16*   RR16  = (h16*)  wsp(d_ws, off, (size_t)BB*512*64*2);
    h16*   CC16  = (h16*)  wsp(d_ws, off, (size_t)BB*512*64*2);
    float* PROW  = (float*)wsp(d_ws, off, (size_t)BB*512*64*4);
    float* PCOL  = (float*)wsp(d_ws, off, (size_t)BB*512*64*4);
    h16* Wq  = (h16*)wsp(d_ws, off, (size_t)256*256*2);
    h16* Wk  = (h16*)wsp(d_ws, off, (size_t)256*256*2);
    h16* Wv  = (h16*)wsp(d_ws, off, (size_t)512*256*2);
    h16* W3  = (h16*)wsp(d_ws, off, (size_t)512*4608*2);
    h16* Wl1 = (h16*)wsp(d_ws, off, (size_t)512*512*2);
    h16* Wl3 = (h16*)wsp(d_ws, off, (size_t)512*4608*2);
    h16* Wpw = (h16*)wsp(d_ws, off, (size_t)256*1024*2);
    h16* Wpj = (h16*)wsp(d_ws, off, (size_t)256*512*2);
    h16* Wpr = (h16*)wsp(d_ws, off, (size_t)512*512*2);
    h16* Wpc = (h16*)wsp(d_ws, off, (size_t)512*512*2);
    float* Bq  = (float*)wsp(d_ws, off, 256*4);
    float* Bk  = (float*)wsp(d_ws, off, 256*4);
    float* Bv  = (float*)wsp(d_ws, off, 512*4);
    float* B3  = (float*)wsp(d_ws, off, 512*4);
    float* Bl1 = (float*)wsp(d_ws, off, 512*4);
    float* Bl3 = (float*)wsp(d_ws, off, 512*4);
    float* Bpw = (float*)wsp(d_ws, off, 256*4);
    float* Bpj = (float*)wsp(d_ws, off, 256*4);
    float* Bpr = (float*)wsp(d_ws, off, 512*4);
    float* Bpc = (float*)wsp(d_ws, off, 512*4);

    auto prep = [&](const float* w, const float* g, const float* b,
                    const float* m, const float* v, float eps,
                    int Co, int Kw, h16* w16, float* bias) {
        int total = Co * Kw;
        prep_weights<<<(total + 255) / 256, 256, 0, stream>>>(w, g, b, m, v, eps, Co, Kw, w16, bias);
    };
    prep(tq_w, tq_g, tq_b, tq_m, tq_v, 1e-5f, 256, 256,  Wq,  Bq);
    prep(tk_w, tk_g, tk_b, tk_m, tk_v, 1e-5f, 256, 256,  Wk,  Bk);
    prep(tv_w, tv_g, tv_b, tv_m, tv_v, 1e-5f, 512, 256,  Wv,  Bv);
    prep(c3_w, c3_g, c3_b, c3_m, c3_v, 1e-3f, 512, 4608, W3,  B3);
    prep(l1_w, l1_g, l1_b, l1_m, l1_v, 1e-5f, 512, 512,  Wl1, Bl1);
    prep(l3_w, l3_g, l3_b, l3_m, l3_v, 1e-5f, 512, 4608, Wl3, Bl3);
    prep(pw_w, pw_g, pw_b, pw_m, pw_v, 1e-5f, 256, 1024, Wpw, Bpw);
    prep(pj_w, pj_g, pj_b, pj_m, pj_v, 1e-5f, 256, 512,  Wpj, Bpj);
    prep(pr_w, pr_g, pr_b, pr_m, pr_v, 1e-5f, 512, 512,  Wpr, Bpr);
    prep(pc_w, pc_g, pc_b, pc_m, pc_v, 1e-5f, 512, 512,  Wpc, Bpc);

    // q,k,v = convbn(x)  (f32 input path)
    gemm1x1_wmma<1,0,0><<<dim3(64, 2, BB), 256, 0, stream>>>(
        Wq, nullptr, x, Bq, nullptr, QK16, nullptr, nullptr, nullptr,
        256, 256, HWP, 512, 0);
    gemm1x1_wmma<1,0,0><<<dim3(64, 2, BB), 256, 0, stream>>>(
        Wk, nullptr, x, Bk, nullptr, QK16, nullptr, nullptr, nullptr,
        256, 256, HWP, 512, 256);
    gemm1x1_wmma<1,0,0><<<dim3(64, 4, BB), 256, 0, stream>>>(
        Wv, nullptr, x, Bv, nullptr, V16, nullptr, nullptr, nullptr,
        512, 256, HWP, 512, 0);

    // axial pooling
    pool_means<<<(BB * 512 * 64 + 255) / 256, 256, 0, stream>>>(QK16, 512, QKROW, QKCOL);
    pool_means<<<(BB * 512 * 64 + 255) / 256, 256, 0, stream>>>(V16, 512, VROW, VCOL);

    // x2 branch: conv3x3(+bn+prelu) -> depthwise fuse -> QKV2[:,512:]
    gemm3x3_wmma<2,0><<<dim3(64, 4, BB), 256, 0, stream>>>(
        W3, V16, B3, c3_a, BR16, nullptr, 512, 512, 512, 0);
    dw_fuse<<<(unsigned)((size_t)BB * 512 * HWP / 256), 256, 0, stream>>>(
        BR16, dc_w, dc_g, dc_b, dc_m, dc_v, dc_a,
        dd_w, dd_g, dd_b, dd_m, dd_v, dd_a,
        bp_g, bp_b, bp_m, bp_v, bp_a,
        QKV2, 1024, 512);

    // x1 branch: prelu(left1) + prelu(left3) -> QKV2[:,0:512]
    gemm1x1_wmma<0,3,0><<<dim3(64, 4, BB), 256, 0, stream>>>(
        Wl1, QK16, nullptr, Bl1, l1_a, T16, nullptr, nullptr, nullptr,
        512, 512, HWP, 512, 0);
    gemm3x3_wmma<3,1><<<dim3(64, 4, BB), 256, 0, stream>>>(
        Wl3, QK16, Bl3, l3_a, QKV2, T16, 512, 512, 1024, 0);

    // pwconv: 1024 -> 256 (f32 out; gets gated at the end)
    gemm1x1_wmma<0,0,2><<<dim3(64, 2, BB), 256, 0, stream>>>(
        Wpw, QKV2, nullptr, Bpw, nullptr, nullptr, PW, nullptr, nullptr,
        256, 1024, HWP, 256, 0);

    // axial attention
    axial_attn<<<dim3(8, BB), 64, 0, stream>>>(QKROW, VROW, pe_rowq, pe_rowk, RR16);
    axial_attn<<<dim3(8, BB), 64, 0, stream>>>(QKCOL, VCOL, pe_colq, pe_colk, CC16);

    // proj_row / proj_col (P=64)
    gemm1x1_wmma<0,0,2><<<dim3(1, 4, BB), 256, 0, stream>>>(
        Wpr, RR16, nullptr, Bpr, nullptr, nullptr, PROW, nullptr, nullptr,
        512, 512, 64, 512, 0);
    gemm1x1_wmma<0,0,2><<<dim3(1, 4, BB), 256, 0, stream>>>(
        Wpc, CC16, nullptr, Bpc, nullptr, nullptr, PCOL, nullptr, nullptr,
        512, 512, 64, 512, 0);

    // xx = relu(v + row + col) -> reuse BR16
    xx_combine<<<(unsigned)((size_t)BB * 512 * HWP / 256), 256, 0, stream>>>(
        V16, PROW, PCOL, BR16);

    // final: proj (512->256) + hard-sigmoid gate * pwconv output -> d_out
    gemm1x1_wmma<0,0,3><<<dim3(64, 2, BB), 256, 0, stream>>>(
        Wpj, BR16, nullptr, Bpj, nullptr, nullptr, (float*)d_out, nullptr, PW,
        256, 512, HWP, 256, 0);
#undef FIN
}
